// HVRAE_40973988004379
// MI455X (gfx1250) — matrix operand order repaired and checked by reading, not compiled, not run on hardware
//
#include <hip/hip_runtime.h>
#include <hip/hip_bf16.h>

#define DEV __device__ __forceinline__

typedef __attribute__((ext_vector_type(16))) _Float16 v16h;
typedef __attribute__((ext_vector_type(8)))  float    v8f;

union V16H { v16h v; _Float16 e[16]; };
union V8F  { v8f  v; float    e[8];  };

DEV float fast_tanh(float x) {
#if __has_builtin(__builtin_amdgcn_tanhf)
  return __builtin_amdgcn_tanhf(x);
#elif __has_builtin(__builtin_amdgcn_tanh_f32)
  return __builtin_amdgcn_tanh_f32(x);
#else
  return tanhf(x);
#endif
}

// 16-bit A-matrix 16x32 element K index for (vgpr v, sub s, lane-half)
DEV int a_k(int v, int s, int half) { return ((v >> 2) << 4) + half * 8 + ((v & 3) << 1) + s; }
// 16-bit B-matrix 32x16 element K index for (vgpr v, sub s, lane-half)
DEV int b_k(int v, int s, int half) { return ((v >> 2) << 4) + half * 8 + ((v & 3) << 1) + s; }

DEV v8f wmma_f16(v16h a, v16h b, v8f c) {
  return __builtin_amdgcn_wmma_f32_16x16x32_f16(false, a, false, b, (short)0, c, false, false);
}

DEV v16h zero16h() {
  V16H r;
  for (int i = 0; i < 16; ++i) r.e[i] = (_Float16)0.0f;
  return r.v;
}

// Build B operand (= W^T slice) from row-major W (N_out rows, `stride` cols).
// Covers output cols [noff, noff+16), K slice [koff, koff+32), zero beyond kmax.
DEV v16h load_wB(const float* W, int stride, int noff, int koff, int kmax, int lane) {
  int n = lane & 15, half = lane >> 4;
  V16H r;
  for (int v = 0; v < 8; ++v)
    for (int s = 0; s < 2; ++s) {
      int k = koff + b_k(v, s, half);
      float val = (k < kmax) ? W[(noff + n) * stride + k] : 0.0f;
      r.e[2 * v + s] = (_Float16)val;
    }
  return r.v;
}

// Build A operand from a row-major LDS tile (16 rows x stride floats).
DEV v16h load_aLDS(const float* buf, int stride, int koff, int kmax, int lane) {
  int m = lane & 15, half = lane >> 4;
  V16H r;
  for (int v = 0; v < 8; ++v)
    for (int s = 0; s < 2; ++s) {
      int k = koff + a_k(v, s, half);
      float val = (k < kmax) ? buf[m * stride + k] : 0.0f;
      r.e[2 * v + s] = (_Float16)val;
    }
  return r.v;
}

// ---------------------------------------------------------------------------
// Kernel 1: front-end.  h = tanh(x@fc_w^T+b); zm/zlv; z = zm+exp(.5 zlv)*eps;
// pre0 = z@enc_wih0^T + enc_bih0.   One wave per 16-row tile of B*T rows.
// ---------------------------------------------------------------------------
__global__ __launch_bounds__(256) void fe_kernel(
    const float* __restrict__ x, const float* __restrict__ eps,
    const float* __restrict__ fc_w, const float* __restrict__ fc_b,
    const float* __restrict__ zm_w, const float* __restrict__ zm_b,
    const float* __restrict__ zlv_w, const float* __restrict__ zlv_b,
    const float* __restrict__ wih0, const float* __restrict__ bih0,
    float* __restrict__ pre0) {
  __shared__ float s_h[8][16][68];
  int lane = threadIdx.x & 31, w = threadIdx.x >> 5;
  int row0 = (blockIdx.x * 8 + w) * 16;
  int n = lane & 15, half = lane >> 4;

  // ---- h = tanh(x @ fc_w^T + fc_b): K=256 in 8 slices, N=64 in 4 tiles ----
  V8F acc[4];
  for (int nt = 0; nt < 4; ++nt)
    for (int v = 0; v < 8; ++v) acc[nt].e[v] = fc_b[nt * 16 + n];

  for (int ks = 0; ks < 8; ++ks) {
    V16H a;
    for (int v = 0; v < 8; ++v)
      for (int s = 0; s < 2; ++s) {
        int k = ks * 32 + a_k(v, s, half);
        a.e[2 * v + s] = (_Float16)x[(row0 + n) * 256 + k];
      }
    for (int nt = 0; nt < 4; ++nt) {
      v16h bw = load_wB(fc_w, 256, nt * 16, ks * 32, 256, lane);
      acc[nt].v = wmma_f16(a.v, bw, acc[nt].v);
    }
  }
  for (int nt = 0; nt < 4; ++nt)
    for (int v = 0; v < 8; ++v)
      s_h[w][v + 8 * half][nt * 16 + n] = fast_tanh(acc[nt].e[v]);

  v16h ah0 = load_aLDS(&s_h[w][0][0], 68, 0, 64, lane);
  v16h ah1 = load_aLDS(&s_h[w][0][0], 68, 32, 64, lane);

  // ---- z_mean / z_log_var (64 -> 64) ----
  V8F zm[4], zlv[4];
  for (int nt = 0; nt < 4; ++nt) {
    for (int v = 0; v < 8; ++v) {
      zm[nt].e[v] = zm_b[nt * 16 + n];
      zlv[nt].e[v] = zlv_b[nt * 16 + n];
    }
    zm[nt].v = wmma_f16(ah0, load_wB(zm_w, 64, nt * 16, 0, 64, lane), zm[nt].v);
    zm[nt].v = wmma_f16(ah1, load_wB(zm_w, 64, nt * 16, 32, 64, lane), zm[nt].v);
    zlv[nt].v = wmma_f16(ah0, load_wB(zlv_w, 64, nt * 16, 0, 64, lane), zlv[nt].v);
    zlv[nt].v = wmma_f16(ah1, load_wB(zlv_w, 64, nt * 16, 32, 64, lane), zlv[nt].v);
  }
  // ---- reparameterize, stage z for next GEMM ----
  for (int nt = 0; nt < 4; ++nt)
    for (int v = 0; v < 8; ++v) {
      int r = row0 + v + 8 * half;
      float z = zm[nt].e[v] + __expf(0.5f * zlv[nt].e[v]) * eps[r * 64 + nt * 16 + n];
      s_h[w][v + 8 * half][nt * 16 + n] = z;
    }
  v16h az0 = load_aLDS(&s_h[w][0][0], 68, 0, 64, lane);
  v16h az1 = load_aLDS(&s_h[w][0][0], 68, 32, 64, lane);

  // ---- pre0 = z @ enc_wih0^T + enc_bih0 (64 -> 16) ----
  V8F p;
  for (int v = 0; v < 8; ++v) p.e[v] = bih0[n];
  p.v = wmma_f16(az0, load_wB(wih0, 64, 0, 0, 64, lane), p.v);
  p.v = wmma_f16(az1, load_wB(wih0, 64, 0, 32, 64, lane), p.v);
  for (int v = 0; v < 8; ++v) pre0[(row0 + v + 8 * half) * 16 + n] = p.e[v];
}

// ---------------------------------------------------------------------------
// Kernel 2: fused, software-pipelined encoder+decoder scan over b.
// wave 0 = encoder (3 Elman layers), wave 1 = decoder (3 Elman layers),
// chunk-wise hand-off (CH steps) through a double-buffered LDS ring.
// Decoder trails encoder by one chunk -> the two 8192-step serial chains
// overlap almost completely (~2x on the latency wall).
// ---------------------------------------------------------------------------
__global__ __launch_bounds__(64) void scan_kernel(
    const float* __restrict__ pre0,
    const float* __restrict__ e_whh0, const float* __restrict__ e_bhh0,
    const float* __restrict__ e_wih1, const float* __restrict__ e_bih1,
    const float* __restrict__ e_whh1, const float* __restrict__ e_bhh1,
    const float* __restrict__ e_wih2, const float* __restrict__ e_bih2,
    const float* __restrict__ e_whh2, const float* __restrict__ e_bhh2,
    const float* __restrict__ d_wih0, const float* __restrict__ d_bih0,
    const float* __restrict__ d_whh0, const float* __restrict__ d_bhh0,
    const float* __restrict__ d_wih1, const float* __restrict__ d_bih1,
    const float* __restrict__ d_whh1, const float* __restrict__ d_bhh1,
    const float* __restrict__ d_wih2, const float* __restrict__ d_bih2,
    const float* __restrict__ d_whh2, const float* __restrict__ d_bhh2,
    float* __restrict__ dec_out, int nB) {
  constexpr int CH = 64;
  __shared__ float encbuf[2][CH][16];  // enc layer-2 row t=9, per step in chunk
  __shared__ float sb[2][16][20];      // per-wave D->A conversion tile
  const int lane = threadIdx.x & 31;
  const int wid = threadIdx.x >> 5;
  const int n = lane & 15, half = lane >> 4;
  float* mysb = &sb[wid][0][0];
  const int NC = nB / CH;

  // role-dependent weights (B layout, f16) and fused bias terms
  v16h W0, W1, W2, W3, W4, W5;
  float cb0, cb1, cb2;
  if (wid == 0) {  // encoder: whh0, wih1, whh1, wih2, whh2
    W0 = load_wB(e_whh0, 16, 0, 0, 16, lane);
    W1 = load_wB(e_wih1, 16, 0, 0, 16, lane);
    W2 = load_wB(e_whh1, 16, 0, 0, 16, lane);
    W3 = load_wB(e_wih2, 16, 0, 0, 16, lane);
    W4 = load_wB(e_whh2, 16, 0, 0, 16, lane);
    W5 = zero16h();
    cb0 = e_bhh0[n];  // bih0 already folded into pre0
    cb1 = e_bih1[n] + e_bhh1[n];
    cb2 = e_bih2[n] + e_bhh2[n];
  } else {  // decoder: wih0, whh0, wih1, whh1, wih2, whh2
    W0 = load_wB(d_wih0, 16, 0, 0, 16, lane);
    W1 = load_wB(d_whh0, 16, 0, 0, 16, lane);
    W2 = load_wB(d_wih1, 16, 0, 0, 16, lane);
    W3 = load_wB(d_whh1, 16, 0, 0, 16, lane);
    W4 = load_wB(d_wih2, 16, 0, 0, 16, lane);
    W5 = load_wB(d_whh2, 16, 0, 0, 16, lane);
    cb0 = d_bih0[n] + d_bhh0[n];
    cb1 = d_bih1[n] + d_bhh1[n];
    cb2 = d_bih2[n] + d_bhh2[n];
  }
  v16h h0 = zero16h(), h1 = zero16h(), h2 = zero16h();

  for (int c = 0; c <= NC; ++c) {
    if (wid == 0) {
      if (c < NC) {  // -------- encoder chunk c --------
        int bb = c & 1;
        for (int j = 0; j < CH; ++j) {
          int b = c * CH + j;
          __builtin_prefetch(&pre0[(b + 4) * 160], 0, 1);
          // layer 0: tanh(pre0_b + h0 @ Whh0^T + bhh0)
          V8F c0;
          for (int v = 0; v < 8; ++v) {
            int t = v + 8 * half;
            c0.e[v] = cb0 + ((t < 10) ? pre0[(b * 10 + t) * 16 + n] : 0.0f);
          }
          c0.v = wmma_f16(h0, W0, c0.v);
          for (int v = 0; v < 8; ++v) mysb[(v + 8 * half) * 20 + n] = fast_tanh(c0.e[v]);
          h0 = load_aLDS(mysb, 20, 0, 16, lane);
          // layer 1
          V8F c1;
          for (int v = 0; v < 8; ++v) c1.e[v] = cb1;
          c1.v = wmma_f16(h0, W1, c1.v);
          c1.v = wmma_f16(h1, W2, c1.v);
          for (int v = 0; v < 8; ++v) mysb[(v + 8 * half) * 20 + n] = fast_tanh(c1.e[v]);
          h1 = load_aLDS(mysb, 20, 0, 16, lane);
          // layer 2
          V8F c2;
          for (int v = 0; v < 8; ++v) c2.e[v] = cb2;
          c2.v = wmma_f16(h1, W3, c2.v);
          c2.v = wmma_f16(h2, W4, c2.v);
          for (int v = 0; v < 8; ++v) c2.e[v] = fast_tanh(c2.e[v]);
          if (half == 1) encbuf[bb][j][n] = c2.e[1];  // row M = 9 == t = T-1
          for (int v = 0; v < 8; ++v) mysb[(v + 8 * half) * 20 + n] = c2.e[v];
          h2 = load_aLDS(mysb, 20, 0, 16, lane);
        }
      }
    } else {
      if (c >= 1) {  // -------- decoder chunk c-1 --------
        int bb = (c - 1) & 1;
        for (int j = 0; j < CH; ++j) {
          int b = (c - 1) * CH + j;
          // broadcast encoder latent to every row of the A tile
          V16H ain;
          for (int v = 0; v < 8; ++v)
            for (int s = 0; s < 2; ++s) {
              int k = a_k(v, s, half);
              ain.e[2 * v + s] =
                  (k < 16) ? (_Float16)encbuf[bb][j][k] : (_Float16)0.0f;
            }
          // layer 0
          V8F c0;
          for (int v = 0; v < 8; ++v) c0.e[v] = cb0;
          c0.v = wmma_f16(ain.v, W0, c0.v);
          c0.v = wmma_f16(h0, W1, c0.v);
          for (int v = 0; v < 8; ++v) mysb[(v + 8 * half) * 20 + n] = fast_tanh(c0.e[v]);
          h0 = load_aLDS(mysb, 20, 0, 16, lane);
          // layer 1
          V8F c1;
          for (int v = 0; v < 8; ++v) c1.e[v] = cb1;
          c1.v = wmma_f16(h0, W2, c1.v);
          c1.v = wmma_f16(h1, W3, c1.v);
          for (int v = 0; v < 8; ++v) mysb[(v + 8 * half) * 20 + n] = fast_tanh(c1.e[v]);
          h1 = load_aLDS(mysb, 20, 0, 16, lane);
          // layer 2
          V8F c2;
          for (int v = 0; v < 8; ++v) c2.e[v] = cb2;
          c2.v = wmma_f16(h1, W4, c2.v);
          c2.v = wmma_f16(h2, W5, c2.v);
          for (int v = 0; v < 8; ++v) c2.e[v] = fast_tanh(c2.e[v]);
          for (int v = 0; v < 8; ++v) {
            int t = v + 8 * half;
            if (t < 10) dec_out[(b * 10 + t) * 16 + n] = c2.e[v];
          }
          for (int v = 0; v < 8; ++v) mysb[(v + 8 * half) * 20 + n] = c2.e[v];
          h2 = load_aLDS(mysb, 20, 0, 16, lane);
        }
      }
    }
    __syncthreads();  // publish encbuf chunk c, retire chunk c-1
  }
}

// ---------------------------------------------------------------------------
// Kernel 3: back-end. time-reverse, xl = tanh(d @ xl_w^T + b) (16->64),
// pm/pl packed as one 64->8 GEMM, broadcast over P=64.
// ---------------------------------------------------------------------------
__global__ __launch_bounds__(256) void be_kernel(
    const float* __restrict__ dec_out,
    const float* __restrict__ xl_w, const float* __restrict__ xl_b,
    const float* __restrict__ pm_w, const float* __restrict__ pm_b,
    const float* __restrict__ pl_w, const float* __restrict__ pl_b,
    float* __restrict__ out) {
  __shared__ float s_xl[8][16][68];
  __shared__ __align__(16) float s_xz[8][16][8];
  int lane = threadIdx.x & 31, w = threadIdx.x >> 5;
  int row0 = (blockIdx.x * 8 + w) * 16;
  int n = lane & 15, half = lane >> 4;

  // A = time-reversed decoder output (16 rows x K=16, padded to K=32)
  V16H a;
  for (int v = 0; v < 8; ++v)
    for (int s = 0; s < 2; ++s) {
      int k = a_k(v, s, half);
      float val = 0.0f;
      if (k < 16) {
        int r = row0 + n;
        int bb = r / 10, t = r % 10;
        val = dec_out[(bb * 10 + (9 - t)) * 16 + k];
      }
      a.e[2 * v + s] = (_Float16)val;
    }
  for (int nt = 0; nt < 4; ++nt) {
    V8F c;
    for (int v = 0; v < 8; ++v) c.e[v] = xl_b[nt * 16 + n];
    c.v = wmma_f16(a.v, load_wB(xl_w, 16, nt * 16, 0, 16, lane), c.v);
    for (int v = 0; v < 8; ++v) s_xl[w][v + 8 * half][nt * 16 + n] = fast_tanh(c.e[v]);
  }
  v16h ax0 = load_aLDS(&s_xl[w][0][0], 68, 0, 64, lane);
  v16h ax1 = load_aLDS(&s_xl[w][0][0], 68, 32, 64, lane);

  V8F cz;
  for (int v = 0; v < 8; ++v)
    cz.e[v] = (n < 4) ? pm_b[n] : ((n < 8) ? pl_b[n - 4] : 0.0f);
  for (int ks = 0; ks < 2; ++ks) {
    V16H bw;  // packed [pm_w ; pl_w] as 8 used output columns
    for (int v = 0; v < 8; ++v)
      for (int s = 0; s < 2; ++s) {
        int k = ks * 32 + b_k(v, s, half);
        float val = 0.0f;
        if (k < 64) {
          if (n < 4) val = pm_w[n * 64 + k];
          else if (n < 8) val = pl_w[(n - 4) * 64 + k];
        }
        bw.e[2 * v + s] = (_Float16)val;
      }
    cz.v = wmma_f16(ks ? ax1 : ax0, bw.v, cz.v);
  }
  if (n < 8)
    for (int v = 0; v < 8; ++v) s_xz[w][v + 8 * half][n] = cz.e[v];

  // broadcast over P=64 with coalesced b128 stores: 16 rows * 512 floats
  long base = (long)row0 * 512;
  for (int i = lane * 4; i < 16 * 512; i += 32 * 4) {
    int row = i >> 9;
    int j = i & 7;  // 0 or 4
    float4 val = *(const float4*)&s_xz[w][row][j];
    *(float4*)&out[base + i] = val;
  }
}

extern "C" void kernel_launch(void* const* d_in, const int* in_sizes, int n_in,
                              void* d_out, int out_size, void* d_ws, size_t ws_size,
                              hipStream_t stream) {
  const float* x     = (const float*)d_in[0];
  const float* eps   = (const float*)d_in[1];
  const float* fc_w  = (const float*)d_in[2];
  const float* fc_b  = (const float*)d_in[3];
  const float* zm_w  = (const float*)d_in[4];
  const float* zm_b  = (const float*)d_in[5];
  const float* zlv_w = (const float*)d_in[6];
  const float* zlv_b = (const float*)d_in[7];
  const float* e_wih0 = (const float*)d_in[8];
  const float* e_bih0 = (const float*)d_in[9];
  const float* e_whh0 = (const float*)d_in[10];
  const float* e_bhh0 = (const float*)d_in[11];
  const float* e_wih1 = (const float*)d_in[12];
  const float* e_bih1 = (const float*)d_in[13];
  const float* e_whh1 = (const float*)d_in[14];
  const float* e_bhh1 = (const float*)d_in[15];
  const float* e_wih2 = (const float*)d_in[16];
  const float* e_bih2 = (const float*)d_in[17];
  const float* e_whh2 = (const float*)d_in[18];
  const float* e_bhh2 = (const float*)d_in[19];
  const float* d_wih0 = (const float*)d_in[20];
  const float* d_bih0 = (const float*)d_in[21];
  const float* d_whh0 = (const float*)d_in[22];
  const float* d_bhh0 = (const float*)d_in[23];
  const float* d_wih1 = (const float*)d_in[24];
  const float* d_bih1 = (const float*)d_in[25];
  const float* d_whh1 = (const float*)d_in[26];
  const float* d_bhh1 = (const float*)d_in[27];
  const float* d_wih2 = (const float*)d_in[28];
  const float* d_bih2 = (const float*)d_in[29];
  const float* d_whh2 = (const float*)d_in[30];
  const float* d_bhh2 = (const float*)d_in[31];
  const float* xl_w  = (const float*)d_in[32];
  const float* xl_b  = (const float*)d_in[33];
  const float* pm_w  = (const float*)d_in[34];
  const float* pm_b  = (const float*)d_in[35];
  const float* pl_w  = (const float*)d_in[36];
  const float* pl_b  = (const float*)d_in[37];

  const int Bn = 8192, T = 10;
  const int R = Bn * T;  // 81920 rows

  float* ws = (float*)d_ws;
  float* pre0    = ws;                      // R * 16 floats
  float* dec_out = pre0 + (size_t)R * 16;   // R * 16 floats

  fe_kernel<<<R / 128, 256, 0, stream>>>(x, eps, fc_w, fc_b, zm_w, zm_b, zlv_w,
                                         zlv_b, e_wih0, e_bih0, pre0);
  scan_kernel<<<1, 64, 0, stream>>>(
      pre0, e_whh0, e_bhh0, e_wih1, e_bih1, e_whh1, e_bhh1, e_wih2, e_bih2,
      e_whh2, e_bhh2, d_wih0, d_bih0, d_whh0, d_bhh0, d_wih1, d_bih1, d_whh1,
      d_bhh1, d_wih2, d_bih2, d_whh2, d_bhh2, dec_out, Bn);
  be_kernel<<<R / 128, 256, 0, stream>>>(dec_out, xl_w, xl_b, pm_w, pm_b, pl_w,
                                         pl_b, (float*)d_out);
}
